// SkipGRU_82274393523068
// MI455X (gfx1250) — matrix operand order repaired and checked by reading, not compiled, not run on hardware
//
#include <hip/hip_runtime.h>

#define B_ 64
#define T_ 512
#define N_ 1024
#define H_ 1024

typedef __attribute__((ext_vector_type(16))) __bf16 v16bf;
typedef __attribute__((ext_vector_type(8)))  float  v8f;

union Frag { v16bf v; uint4 q[2]; unsigned u[8]; };

__device__ inline unsigned short bf16rne(float f) {
    unsigned u = __builtin_bit_cast(unsigned, f);
    return (unsigned short)((u + 0x7FFFu + ((u >> 16) & 1u)) >> 16);
}
__device__ inline unsigned bf16pair(float f0, float f1) {
    return ((unsigned)bf16rne(f1) << 16) | bf16rne(f0);
}

// 32-chunk pipelined WMMA region. a/b advance 64 uint4 (1KB) per chunk; fully
// unrolled so all loads use immediate offsets and double-buffering overlaps
// loads with v_wmma issue.
__device__ inline v8f mm32(v8f acc, const uint4* __restrict__ a, const uint4* __restrict__ b) {
    Frag fa0, fb0, fa1, fb1;
    fa0.q[0] = a[0]; fa0.q[1] = a[1];
    fb0.q[0] = b[0]; fb0.q[1] = b[1];
#pragma unroll
    for (int kc = 0; kc < 32; ++kc) {
        if (kc + 1 < 32) {
            const uint4* an = a + (kc + 1) * 64;
            const uint4* bn = b + (kc + 1) * 64;
            if (kc & 1) {
                fa0.q[0] = an[0]; fa0.q[1] = an[1];
                fb0.q[0] = bn[0]; fb0.q[1] = bn[1];
            } else {
                fa1.q[0] = an[0]; fa1.q[1] = an[1];
                fb1.q[0] = bn[0]; fb1.q[1] = bn[1];
            }
        }
        if (kc & 1)
            acc = __builtin_amdgcn_wmma_f32_16x16x32_bf16(false, fa1.v, false, fb1.v,
                                                          (short)0, acc, false, false);
        else
            acc = __builtin_amdgcn_wmma_f32_16x16x32_bf16(false, fa0.v, false, fb0.v,
                                                          (short)0, acc, false, false);
    }
    return acc;
}

// ---- Wpack: [g:3][ct:64][kc:64][lane:32][v:8] dwords (bf16 pairs), B-fragment layout
// B 32x16 layout: N = lane%16, K0 = (v%4)*2 + (lane/16)*8 + (v/4)*16
__global__ void skipgru_wpack(const float* __restrict__ Wih,
                              const float* __restrict__ Whh,
                              unsigned* __restrict__ Wp) {
    long i = (long)blockIdx.x * blockDim.x + threadIdx.x;  // 3*64*64*32*8 dwords
    int v    = (int)(i & 7);
    int lane = (int)((i >> 3) & 31);
    int kc   = (int)((i >> 8) & 63);
    int ct   = (int)((i >> 14) & 63);
    int g    = (int)(i >> 20);
    int row  = g * 1024 + ct * 16 + (lane & 15);
    int k    = kc * 32 + (v & 3) * 2 + (lane >> 4) * 8 + (v >> 2) * 16;
    float f0, f1;
    if (k < N_) { f0 = Wih[(long)row * N_ + k];        f1 = Wih[(long)row * N_ + k + 1]; }
    else        { f0 = Whh[(long)row * H_ + (k - N_)]; f1 = Whh[(long)row * H_ + (k - N_) + 1]; }
    Wp[i] = bf16pair(f0, f1);
}

// ---- Xpack: [t:512][rt:4][kc:32][lane:32][v:8] dwords, A-fragment layout
// A 16x32 layout: M = lane%16, K0 = (v%4)*2 + (lane/16)*8 + (v/4)*16
__global__ void skipgru_xpack(const float* __restrict__ x, unsigned* __restrict__ Xp) {
    long i = (long)blockIdx.x * blockDim.x + threadIdx.x;  // 512*4*32*32*8 dwords
    int v    = (int)(i & 7);
    int lane = (int)((i >> 3) & 31);
    int kc   = (int)((i >> 8) & 31);
    int rt   = (int)((i >> 13) & 3);
    int t    = (int)(i >> 15);
    int b    = rt * 16 + (lane & 15);
    int k    = kc * 32 + (v & 3) * 2 + (lane >> 4) * 8 + (v >> 2) * 16;
    const float* src = x + ((long)b * T_ + t) * (long)N_ + k;
    Xp[i] = bf16pair(src[0], src[1]);
}

__global__ void skipgru_zero(unsigned* __restrict__ p) {
    p[(long)blockIdx.x * blockDim.x + threadIdx.x] = 0u;
}

// combined biases: cb[0..1023]=r, [1024..2047]=z, [2048..]=xn, [3072..]=hn
__global__ void skipgru_cbias(const float* __restrict__ b_ih, const float* __restrict__ b_hh,
                              float* __restrict__ cb) {
    int i = blockIdx.x * blockDim.x + threadIdx.x;  // 0..4095
    int g = i >> 10, c = i & 1023;
    float f;
    if (g == 0)      f = b_ih[c]          + b_hh[c];
    else if (g == 1) f = b_ih[H_ + c]     + b_hh[H_ + c];
    else if (g == 2) f = b_ih[2 * H_ + c];
    else             f = b_hh[2 * H_ + c];
    cb[i] = f;
}

// One recurrent step. Grid: 64 WGs (16 cols each) x 512 threads (16 waves).
// wave = 4*part + rowtile; parts: 0=r, 1=z (x+h regions), 2=xn (x only), 3=hn (h only)
__global__ __launch_bounds__(512)
void skipgru_step(const uint4* __restrict__ Xp_t, const uint4* __restrict__ Wp,
                  const uint4* __restrict__ Hp_cur, unsigned short* __restrict__ Hp_nxt,
                  const float* __restrict__ mix, const float* __restrict__ cb,
                  const float* __restrict__ h_cur, float* __restrict__ h_nxt,
                  float* __restrict__ out_h, float* __restrict__ out_o, int t) {
    __shared__ float pre[4][B_][16];

    const int tid  = threadIdx.x;
    const int lane = tid & 31;
    const int wave = tid >> 5;
    const int rt   = wave & 3;     // batch row tile
    const int part = wave >> 2;    // 0..3
    const int ct   = blockIdx.x;   // column tile (16 cols)
    const int g    = (part < 2) ? part : 2;

    // per-wave fragment base pointers (uint4 units); chunk stride = 64 uint4
    const uint4* ax = Xp_t   + (long)rt * 2048 + lane * 2;
    const uint4* ah = Hp_cur + (long)rt * 2048 + lane * 2;
    const uint4* bq = Wp + (long)(g * 64 + ct) * 4096 + lane * 2;

    v8f acc = {};
    if (part < 2) {
        acc = mm32(acc, ax, bq);          // x region, chunks 0..31
        acc = mm32(acc, ah, bq + 2048);   // h region, chunks 32..63
    } else if (part == 2) {
        acc = mm32(acc, ax, bq);          // xn: x region only
    } else {
        acc = mm32(acc, ah, bq + 2048);   // hn: h region only
    }

    // C layout: VGPR v -> row v (lanes 0-15) / v+8 (lanes 16-31), col = lane%16
    const int nloc = lane & 15;
    const int lhi  = lane >> 4;
#pragma unroll
    for (int v = 0; v < 8; ++v) {
        pre[part][rt * 16 + v + lhi * 8][nloc] = acc[v];
    }
    __syncthreads();

    const int wgcol = ct * 16;
    for (int idx = tid; idx < B_ * 16; idx += 512) {
        int b  = idx >> 4;
        int c  = idx & 15;
        int gc = wgcol + c;
        float rp = pre[0][b][c] + cb[gc];
        float zp = pre[1][b][c] + cb[1024 + gc];
        float xn = pre[2][b][c] + cb[2048 + gc];
        float hn = pre[3][b][c] + cb[3072 + gc];
        float r = 1.0f / (1.0f + __expf(-rp));
        float z = 1.0f / (1.0f + __expf(-zp));
        float n = tanhf(xn + r * hn);
        float hp = h_cur[(long)b * H_ + gc];
        float hnew = (1.0f - z) * n + z * hp;
        float m = mix[(long)b * T_ + t];
        float hm = hnew * m + hp * (1.0f - m);
        h_nxt[(long)b * H_ + gc] = hm;
        out_h[((long)b * T_ + t) * H_ + gc] = hm;
        if (t == T_ - 1) out_o[(long)b * H_ + gc] = hm;

        // write next-step A fragment element (bf16) into packed layout
        int kcH  = gc >> 5;
        int kk   = gc & 31;
        int bit0 = kk & 1;
        int rem  = kk >> 1;                       // (v%4) + lhi*4 + (v/4)*8
        int v    = (rem & 3) | (((rem >> 3) & 1) << 2);
        int lh   = (rem >> 2) & 1;
        int ln   = (b & 15) | (lh << 4);
        long di  = (((long)(b >> 4) * 32 + kcH) * 32 + ln) * 8 + v;
        Hp_nxt[di * 2 + bit0] = bf16rne(hm);
    }
}

extern "C" void kernel_launch(void* const* d_in, const int* in_sizes, int n_in,
                              void* d_out, int out_size, void* d_ws, size_t ws_size,
                              hipStream_t stream) {
    const float* x    = (const float*)d_in[0];
    const float* mix  = (const float*)d_in[1];
    const float* W_ih = (const float*)d_in[2];
    const float* W_hh = (const float*)d_in[3];
    const float* b_ih = (const float*)d_in[4];
    const float* b_hh = (const float*)d_in[5];

    float* out_h = (float*)d_out;
    float* out_o = out_h + (size_t)B_ * T_ * H_;

    // workspace layout (bytes)
    char* ws = (char*)d_ws;
    const size_t WP_BYTES = (size_t)3 * 64 * 64 * 32 * 8 * 4;       // 12,582,912
    const size_t XP_BYTES = (size_t)512 * 4 * 32 * 32 * 8 * 4;      // 67,108,864
    const size_t HF_BYTES = (size_t)B_ * H_ * 4;                    // 262,144
    const size_t HP_BYTES = (size_t)4 * 32 * 32 * 8 * 4;            // 131,072

    unsigned* Wp_u = (unsigned*)ws;
    unsigned* Xp_u = (unsigned*)(ws + WP_BYTES);
    float*    h0   = (float*)(ws + WP_BYTES + XP_BYTES);
    float*    h1   = (float*)(ws + WP_BYTES + XP_BYTES + HF_BYTES);
    unsigned* Hp0u = (unsigned*)(ws + WP_BYTES + XP_BYTES + 2 * HF_BYTES);
    unsigned* Hp1u = (unsigned*)(ws + WP_BYTES + XP_BYTES + 2 * HF_BYTES + HP_BYTES);
    float*    cb   = (float*)(ws + WP_BYTES + XP_BYTES + 2 * HF_BYTES + 2 * HP_BYTES);

    skipgru_wpack<<<(int)(WP_BYTES / 4 / 256), 256, 0, stream>>>(W_ih, W_hh, Wp_u);
    skipgru_xpack<<<(int)(XP_BYTES / 4 / 256), 256, 0, stream>>>(x, Xp_u);
    // zero h0,h1,Hp0,Hp1 region (contiguous)
    skipgru_zero<<<(int)((2 * HF_BYTES + 2 * HP_BYTES) / 4 / 256), 256, 0, stream>>>((unsigned*)h0);
    skipgru_cbias<<<16, 256, 0, stream>>>(b_ih, b_hh, cb);

    float*    hc  = h0;   float*    hnp = h1;
    unsigned* Hpc = Hp0u; unsigned* Hpn = Hp1u;
    for (int t = 0; t < T_; ++t) {
        const uint4* Xp_t = (const uint4*)(Xp_u + (size_t)t * 4 * 32 * 32 * 8);
        skipgru_step<<<H_ / 16, 512, 0, stream>>>(
            Xp_t, (const uint4*)Wp_u, (const uint4*)Hpc,
            (unsigned short*)Hpn, mix, cb, hc, hnp, out_h, out_o, t);
        float* tf = hc; hc = hnp; hnp = tf;
        unsigned* tu = Hpc; Hpc = Hpn; Hpn = tu;
    }
}
